// MoE_32203664785677
// MI455X (gfx1250) — matrix operand order repaired and checked by reading, not compiled, not run on hardware
//
#include <hip/hip_runtime.h>
#include <hip/hip_bf16.h>

// ---------------------------------------------------------------------------
// MoE (top-2 of 8 routed experts + shared expert), MI455X / gfx1250.
// Compute-bound -> bf16 WMMA (16x16x32, f32 accum), operands pre-converted to
// bf16 in a one-shot streaming pass. v3: 32-token tiles (2 A-frags/wave, so
// each B fragment feeds 2 WMMAs -> half the L2 traffic) + explicit B-fragment
// double buffering so loads for step nt+1 are in flight while WMMAs for nt
// execute (no more s_wait_loadcnt 0x0 in front of every WMMA).
// ---------------------------------------------------------------------------

#define T_TOK 8192
#define DIM   1024
#define NEXP  8
#define INTER 1024
#define TM    32     // tokens per block tile
#define XS_LD 1032   // padded bf16 row stride (2064B = 129*16B -> conflict-free b128 LDS loads)

typedef __attribute__((ext_vector_type(16))) __bf16 v16bf;
typedef __attribute__((ext_vector_type(8)))  float  v8f;

union BFrag {
  v16bf v;
  unsigned short s[16];
  uint4 q[2];
};

__device__ __forceinline__ unsigned short f2bf(float f) {
  unsigned int u = __float_as_uint(f);
  u += 0x7FFFu + ((u >> 16) & 1u);   // round-to-nearest-even
  return (unsigned short)(u >> 16);
}

// A fragment (16x32 bf16, ISA 7.12.2): lane holds row m=lane%16,
// K chunks {kA..kA+7} and {kA+16..kA+23}, where kA = k0 + (lane<16 ? 0 : 8).
__device__ __forceinline__ void load_a_lds(BFrag& f, const unsigned short* row, int kA) {
  f.q[0] = *(const uint4*)(row + kA);
  f.q[1] = *(const uint4*)(row + kA + 16);
}

// B fragment (32x16 bf16): lane holds col n=lane%16, K chunks {kB..kB+7} and
// {kB+16..kB+23}, kB = k0 + (lane<16 ? 0 : 8). Source is pre-converted row-major bf16 weights.
__device__ __forceinline__ void load_b_bf16(BFrag& f, const unsigned short* p) {
  f.q[0] = *(const uint4*)(p);
  f.q[1] = *(const uint4*)(p + 16);
}

__device__ __forceinline__ v8f wmma_bf16(const BFrag& a, const BFrag& b, v8f c) {
  return __builtin_amdgcn_wmma_f32_16x16x32_bf16(false, a.v, false, b.v,
                                                 (short)0, c, false, false);
}

// ---------------------------------------------------------------------------
// Kernel 0: streaming f32 -> bf16 conversion (bandwidth bound, one pass).
// ---------------------------------------------------------------------------
__global__ __launch_bounds__(256) void cvt_bf16_kernel(
    const float* __restrict__ src, unsigned short* __restrict__ dst, long n4)
{
  long i = (long)blockIdx.x * blockDim.x + threadIdx.x;
  const long stride = (long)gridDim.x * blockDim.x;
  for (; i < n4; i += stride) {
    float4 f = ((const float4*)src)[i];
    ushort4 u;
    u.x = f2bf(f.x); u.y = f2bf(f.y); u.z = f2bf(f.z); u.w = f2bf(f.w);
    ((ushort4*)dst)[i] = u;
  }
}

// ---------------------------------------------------------------------------
// Kernel 1: gating. One wave32 per token. gate_w staged in LDS (32KB).
// ---------------------------------------------------------------------------
__global__ __launch_bounds__(256) void moe_gate_kernel(
    const float* __restrict__ x, const float* __restrict__ gw,
    const float* __restrict__ gb, int* __restrict__ counts,
    int* __restrict__ tidx, float* __restrict__ tw)
{
  __shared__ float GW[NEXP][DIM];
  const int tid = threadIdx.x;
  for (int j = tid; j < NEXP * DIM; j += 256)
    GW[j >> 10][j & (DIM - 1)] = gw[j];
  __syncthreads();

  const int lane = tid & 31;
  const int t = blockIdx.x * 8 + (tid >> 5);

  float lg[NEXP];
#pragma unroll
  for (int e = 0; e < NEXP; ++e) lg[e] = 0.f;

  const float* xr = x + (size_t)t * DIM;
  for (int k = lane; k < DIM; k += 32) {
    float xv = xr[k];
#pragma unroll
    for (int e = 0; e < NEXP; ++e) lg[e] = fmaf(xv, GW[e][k], lg[e]);
  }
#pragma unroll
  for (int e = 0; e < NEXP; ++e) {
#pragma unroll
    for (int off = 16; off > 0; off >>= 1)
      lg[e] += __shfl_xor(lg[e], off, 32);
  }

  if (lane == 0) {
    float mx = lg[0];
#pragma unroll
    for (int e = 1; e < NEXP; ++e) mx = fmaxf(mx, lg[e]);
    float p[NEXP], s = 0.f;
#pragma unroll
    for (int e = 0; e < NEXP; ++e) { p[e] = __expf(lg[e] - mx); s += p[e]; }
    float inv = 1.f / s;
    float sc[NEXP], sb[NEXP];
#pragma unroll
    for (int e = 0; e < NEXP; ++e) { sc[e] = p[e] * inv; sb[e] = sc[e] + gb[e]; }

    // top-2 on biased scores; earliest index wins ties (matches lax.top_k)
    int i0 = 0; float b0 = sb[0];
#pragma unroll
    for (int e = 1; e < NEXP; ++e) if (sb[e] > b0) { b0 = sb[e]; i0 = e; }
    int i1 = -1; float b1 = -3.4e38f;
#pragma unroll
    for (int e = 0; e < NEXP; ++e)
      if (e != i0 && sb[e] > b1) { b1 = sb[e]; i1 = e; }

    int p0 = atomicAdd(&counts[i0], 1);
    tidx[(size_t)i0 * T_TOK + p0] = t;
    tw  [(size_t)i0 * T_TOK + p0] = sc[i0];
    int p1 = atomicAdd(&counts[i1], 1);
    tidx[(size_t)i1 * T_TOK + p1] = t;
    tw  [(size_t)i1 * T_TOK + p1] = sc[i1];
  }
}

// ---------------------------------------------------------------------------
// Kernel 2: fused SwiGLU expert on bf16. 32-token tile, 16 waves (512 thr),
// each wave owns a 64-wide N slice and 2 A fragments (rows 0-15 / 16-31).
//   Phase A: H = silu(X@W1^T) * (X@W3^T)   (H in LDS as bf16)
//   Phase B: Y += wgt * (H @ W2^T), scatter via global_atomic_add_f32
// B fragments are double-buffered: nt+1 loads issue before nt's WMMAs.
// ---------------------------------------------------------------------------
__global__ __launch_bounds__(512) void moe_expert_kernel(
    const unsigned short* __restrict__ xb,
    const unsigned short* __restrict__ W1, const unsigned short* __restrict__ W3,
    const unsigned short* __restrict__ W2,
    size_t stride13, size_t stride2, int ld2,
    const int* __restrict__ counts, const int* __restrict__ tidx,
    const float* __restrict__ tw, float* __restrict__ out, int routed)
{
  extern __shared__ unsigned short smem[];
  unsigned short* Xs = smem;                       // [TM][XS_LD] bf16 x rows
  unsigned short* Hs = smem + TM * XS_LD;          // [TM][XS_LD] bf16 H
  int*   toks = (int*)(smem + 2 * TM * XS_LD);
  float* wgts = (float*)(toks + TM);

  const int e = blockIdx.y;
  const int cnt = routed ? counts[e] : T_TOK;
  const int tile0 = blockIdx.x * TM;
  if (tile0 >= cnt) return;

  const unsigned short* W1e = W1 + (size_t)e * stride13;
  const unsigned short* W3e = W3 + (size_t)e * stride13;
  const unsigned short* W2e = W2 + (size_t)e * stride2;

  const int tid = threadIdx.x;
  if (tid < TM) {
    int r = tid;
    if (routed) {
      bool live = (tile0 + r) < cnt;
      int tk = live ? tidx[(size_t)e * T_TOK + tile0 + r] : 0;
      tk = tk < 0 ? 0 : (tk > T_TOK - 1 ? T_TOK - 1 : tk);
      toks[r] = tk;
      wgts[r] = live ? tw[(size_t)e * T_TOK + tile0 + r] : 0.f;
    } else {
      toks[r] = tile0 + r;
      wgts[r] = 1.f;
    }
  }
  __syncthreads();

  // gather bf16 x rows into LDS, 16B chunks (coalesced along D)
  for (int j = tid; j < TM * (DIM / 8); j += 512) {
    int r = j >> 7, c8 = (j & 127) * 8;
    *(uint4*)&Xs[r * XS_LD + c8] = *(const uint4*)&xb[(size_t)toks[r] * DIM + c8];
  }
  __syncthreads();

  const int lane = tid & 31;
  const int n16  = lane & 15;
  const int aoff = (lane < 16) ? 0 : 8;
  const int boff = (lane < 16) ? 0 : 8;
  const int mrow = (lane >> 4) << 3;               // D-frag row base
  const int nbase = (tid >> 5) * 64;               // this wave's 64-wide N slice

  const v8f vzero = {0.f, 0.f, 0.f, 0.f, 0.f, 0.f, 0.f, 0.f};

  // ---- Phase A: gate & up projections -------------------------------------
  v8f acc1[2][4], acc3[2][4];
#pragma unroll
  for (int f = 0; f < 2; ++f)
#pragma unroll
    for (int i = 0; i < 4; ++i) { acc1[f][i] = vzero; acc3[f][i] = vzero; }

  {
    const unsigned short* w1p = W1e + (size_t)(nbase + n16) * DIM + boff;
    const unsigned short* w3p = W3e + (size_t)(nbase + n16) * DIM + boff;
    for (int k0 = 0; k0 < DIM; k0 += 32) {
      BFrag a0, a1, b1[2], b3[2];
      load_a_lds(a0, Xs + (size_t)n16 * XS_LD, k0 + aoff);
      load_a_lds(a1, Xs + (size_t)(n16 + 16) * XS_LD, k0 + aoff);
      load_b_bf16(b1[0], w1p + k0);
      load_b_bf16(b3[0], w3p + k0);
#pragma unroll
      for (int nt = 0; nt < 4; ++nt) {
        const int cur = nt & 1, nxt = cur ^ 1;
        if (nt < 3) {  // prefetch next N-subtile while this one computes
          load_b_bf16(b1[nxt], w1p + (size_t)(nt + 1) * 16 * DIM + k0);
          load_b_bf16(b3[nxt], w3p + (size_t)(nt + 1) * 16 * DIM + k0);
        }
        acc1[0][nt] = wmma_bf16(a0, b1[cur], acc1[0][nt]);
        acc1[1][nt] = wmma_bf16(a1, b1[cur], acc1[1][nt]);
        acc3[0][nt] = wmma_bf16(a0, b3[cur], acc3[0][nt]);
        acc3[1][nt] = wmma_bf16(a1, b3[cur], acc3[1][nt]);
      }
    }
  }

  // silu(g) * u -> Hs (bf16)
#pragma unroll
  for (int f = 0; f < 2; ++f) {
#pragma unroll
    for (int nt = 0; nt < 4; ++nt) {
      const int n = nbase + nt * 16 + n16;
#pragma unroll
      for (int r = 0; r < 8; ++r) {
        float g = acc1[f][nt][r];
        float h = (g / (1.f + __expf(-g))) * acc3[f][nt][r];
        Hs[(size_t)(f * 16 + mrow + r) * XS_LD + n] = f2bf(h);
      }
    }
  }
  __syncthreads();

  // ---- Phase B: down projection + weighted scatter ------------------------
  v8f acc[2][4];
#pragma unroll
  for (int f = 0; f < 2; ++f)
#pragma unroll
    for (int i = 0; i < 4; ++i) acc[f][i] = vzero;

  {
    const unsigned short* w2p = W2e + (size_t)(nbase + n16) * ld2 + boff;
    for (int k0 = 0; k0 < INTER; k0 += 32) {
      BFrag a0, a1, b2[2];
      load_a_lds(a0, Hs + (size_t)n16 * XS_LD, k0 + aoff);
      load_a_lds(a1, Hs + (size_t)(n16 + 16) * XS_LD, k0 + aoff);
      load_b_bf16(b2[0], w2p + k0);
#pragma unroll
      for (int dt = 0; dt < 4; ++dt) {
        const int cur = dt & 1, nxt = cur ^ 1;
        if (dt < 3)
          load_b_bf16(b2[nxt], w2p + (size_t)(dt + 1) * 16 * ld2 + k0);
        acc[0][dt] = wmma_bf16(a0, b2[cur], acc[0][dt]);
        acc[1][dt] = wmma_bf16(a1, b2[cur], acc[1][dt]);
      }
    }
  }

#pragma unroll
  for (int f = 0; f < 2; ++f) {
#pragma unroll
    for (int dt = 0; dt < 4; ++dt) {
      const int d = nbase + dt * 16 + n16;
#pragma unroll
      for (int r = 0; r < 8; ++r) {
        const int m = f * 16 + mrow + r;
        atomicAdd(out + (size_t)toks[m] * DIM + d, wgts[m] * acc[f][dt][r]);
      }
    }
  }
}

// ---------------------------------------------------------------------------
extern "C" void kernel_launch(void* const* d_in, const int* in_sizes, int n_in,
                              void* d_out, int out_size, void* d_ws, size_t ws_size,
                              hipStream_t stream) {
  (void)in_sizes; (void)n_in; (void)ws_size;
  const float* x   = (const float*)d_in[0];
  const float* gw  = (const float*)d_in[1];
  const float* gb  = (const float*)d_in[2];
  const float* w1  = (const float*)d_in[3];
  const float* w3  = (const float*)d_in[4];
  const float* w2  = (const float*)d_in[5];
  const float* sw1 = (const float*)d_in[6];
  const float* sw3 = (const float*)d_in[7];
  const float* sw2 = (const float*)d_in[8];
  float* out = (float*)d_out;

  // ---- workspace layout ----------------------------------------------------
  char* ws = (char*)d_ws;
  int*   counts = (int*)ws;                               ws += 64;
  int*   tidx   = (int*)ws;                               ws += (size_t)NEXP * T_TOK * sizeof(int);
  float* tw     = (float*)ws;                             ws += (size_t)NEXP * T_TOK * sizeof(float);
  unsigned short* xb   = (unsigned short*)ws;             ws += (size_t)T_TOK * DIM * 2;
  unsigned short* w1b  = (unsigned short*)ws;             ws += (size_t)NEXP * INTER * DIM * 2;
  unsigned short* w3b  = (unsigned short*)ws;             ws += (size_t)NEXP * INTER * DIM * 2;
  unsigned short* w2b  = (unsigned short*)ws;             ws += (size_t)NEXP * DIM * INTER * 2;
  unsigned short* sw1b = (unsigned short*)ws;             ws += (size_t)2 * INTER * DIM * 2;
  unsigned short* sw3b = (unsigned short*)ws;             ws += (size_t)2 * INTER * DIM * 2;
  unsigned short* sw2b = (unsigned short*)ws;             ws += (size_t)DIM * 2 * INTER * 2;

  hipMemsetAsync(out, 0, (size_t)out_size * sizeof(float), stream);
  hipMemsetAsync(counts, 0, NEXP * sizeof(int), stream);
  hipMemsetAsync(tw, 0, (size_t)NEXP * T_TOK * sizeof(float), stream);

  // ---- one-shot bf16 conversion (bandwidth bound) --------------------------
  const int CB = 2048;
  cvt_bf16_kernel<<<CB, 256, 0, stream>>>(x,   xb,   (long)T_TOK * DIM / 4);
  cvt_bf16_kernel<<<CB, 256, 0, stream>>>(w1,  w1b,  (long)NEXP * INTER * DIM / 4);
  cvt_bf16_kernel<<<CB, 256, 0, stream>>>(w3,  w3b,  (long)NEXP * INTER * DIM / 4);
  cvt_bf16_kernel<<<CB, 256, 0, stream>>>(w2,  w2b,  (long)NEXP * DIM * INTER / 4);
  cvt_bf16_kernel<<<CB, 256, 0, stream>>>(sw1, sw1b, (long)2 * INTER * DIM / 4);
  cvt_bf16_kernel<<<CB, 256, 0, stream>>>(sw3, sw3b, (long)2 * INTER * DIM / 4);
  cvt_bf16_kernel<<<CB, 256, 0, stream>>>(sw2, sw2b, (long)DIM * 2 * INTER / 4);

  moe_gate_kernel<<<T_TOK / 8, 256, 0, stream>>>(x, gw, gb, counts, tidx, tw);

  const size_t shmem = (size_t)2 * TM * XS_LD * sizeof(unsigned short)
                     + TM * sizeof(int) + TM * sizeof(float);

  // Routed experts: grid covers worst case, early-exit on per-expert count.
  moe_expert_kernel<<<dim3(T_TOK / TM, NEXP), 512, shmem, stream>>>(
      xb, w1b, w3b, w2b,
      (size_t)INTER * DIM, (size_t)DIM * INTER, DIM,
      counts, tidx, tw, out, 1);

  // Shared expert: two 1024-wide slices of the 2048-wide SwiGLU, weight 1.
  moe_expert_kernel<<<dim3(T_TOK / TM, 2), 512, shmem, stream>>>(
      xb, sw1b, sw3b, sw2b,
      (size_t)INTER * DIM, (size_t)INTER, 2 * INTER,
      counts, tidx, tw, out, 0);
}
